// ProteinFeatures_19997367730447
// MI455X (gfx1250) — matrix operand order repaired and checked by reading, not compiled, not run on hardware
//
#include <hip/hip_runtime.h>
#include <hip/hip_bf16.h>
#include <float.h>

typedef __attribute__((ext_vector_type(16))) _Float16 v16h;
typedef __attribute__((ext_vector_type(8)))  _Float16 v8h;
typedef __attribute__((ext_vector_type(8)))  float    v8f;

#define LRES  1024
#define TOPK  30
#define ACTX  25
#define NSIDE 930            // 30 neighbors * 31 side atoms
#define NJ    1108           // 930 + 78 + 100

// ---- output offsets (floats) ----
static const size_t OFF_E     = 0;                       // 1024*30*128
static const size_t OFF_V     = 3932160;                 // 1024*25*128
static const size_t OFF_JN    = 7208960;                 // 1024*25*128
static const size_t OFF_JE    = 10485760;                // 1024*25*25*128
static const size_t OFF_EIDX  = 92405760;                // 1024*30
static const size_t OFF_EIDXJ = 92436480;                // 1024*25
static const size_t OFF_JM    = 92462080;                // 1024*25

__device__ __constant__ int c_side_types[36] =
  {1,0,0,2,0,0,0,0,2,2,4,0,0,0,1,1,2,2,4,0,0,0,0,1,1,1,2,2,0,1,1,2,0,0,0,1};
__device__ __constant__ int c_dna_types[26] =
  {3,2,2,2,0,0,2,0,2,0,2,0,1,0,1,1,0,0,0,1,0,1,2,2,1,0};
// backbone atom ids: N=0, Ca=1, C=2, O=3, Cb=4
__device__ __constant__ int c_pa[24] = {0,2,3,4,1,1,1,1,0,0,0,4,4,3,0,2,3,4,2,3,4,2,3,2};
__device__ __constant__ int c_pb[24] = {0,2,3,4,0,2,3,4,2,3,4,2,3,2,1,1,1,1,0,0,0,4,4,3};

__device__ inline float dist3(const float* a, const float* b) {
  float dx=a[0]-b[0], dy=a[1]-b[1], dz=a[2]-b[2];
  return sqrtf(dx*dx+dy*dy+dz*dz+1e-6f);
}
__device__ inline void rbf16h(float d, _Float16* dst) {
  #pragma unroll
  for (int j=0;j<16;++j){ float mu=2.0f+1.25f*(float)j; float t=(d-mu)*0.8f;
    dst[j]=(_Float16)__expf(-t*t); }
}
__device__ inline void j_atom(int i, int j,
    const float* X, const float* X_m, const float* Y, const float* Y_m,
    const float* Z, const float* Z_m, const int* Z_t, const int* eidx,
    float* p, float* m, int* t) {
  if (j < NSIDE) {
    int knb = j/31, atom = 5 + j%31;
    int nb = eidx[i*TOPK + knb];
    const float* s = X + ((size_t)nb*36 + atom)*3;
    p[0]=s[0]; p[1]=s[1]; p[2]=s[2];
    *m = X_m[nb*36 + atom];
    *t = c_side_types[atom];
  } else if (j < NSIDE + 78) {
    int y = j - NSIDE;
    const float* s = Y + ((size_t)i*78 + y)*3;
    p[0]=s[0]; p[1]=s[1]; p[2]=s[2];
    *m = Y_m[i*78 + y];
    *t = c_dna_types[y % 26];
  } else {
    int z = j - (NSIDE + 78);
    const float* s = Z + (size_t)z*3;
    p[0]=s[0]; p[1]=s[1]; p[2]=s[2];
    *m = Z_m[z];
    *t = Z_t[z];
  }
}

// load 16x32 f16 A-fragment (ISA 7.12.2 layout) from a row-major LDS row
__device__ inline v16h load_a_frag(const _Float16* rowPtr, int ks, int kg) {
  v8h lo = *(const v8h*)(rowPtr + ks*32 + kg*8);
  v8h hi = *(const v8h*)(rowPtr + ks*32 + 16 + kg*8);
  v16h af;
  #pragma unroll
  for (int v=0; v<8; ++v){ af[v]=lo[v]; af[8+v]=hi[v]; }
  return af;
}

// LayerNorm 16x128 tile from LDS staging, 2 lanes per row, streamed float4
__device__ inline void ln_from_lds(const float* sC, const float* gs, const float* gb,
                                   float* outBase, int lane) {
  int row = lane & 15, half = lane >> 4;
  const float4* rp = (const float4*)(sC + row*128 + half*64);
  float s=0.f, q=0.f;
  #pragma unroll
  for (int t=0;t<16;++t){
    float4 xv = rp[t];
    s += xv.x+xv.y+xv.z+xv.w;
    q += xv.x*xv.x+xv.y*xv.y+xv.z*xv.z+xv.w*xv.w;
  }
  s += __shfl_xor(s,16,32); q += __shfl_xor(q,16,32);
  float mean = s*(1.0f/128.0f);
  float var  = q*(1.0f/128.0f) - mean*mean;
  float inv  = rsqrtf(var + 1e-5f);
  const float4* gp = (const float4*)(gs + half*64);
  const float4* bp = (const float4*)(gb + half*64);
  float4* op = (float4*)(outBase + (size_t)row*128 + half*64);
  #pragma unroll
  for (int t=0;t<16;++t){
    float4 xv = rp[t], gv = gp[t], bv = bp[t], r;
    r.x=(xv.x-mean)*inv*gv.x+bv.x;
    r.y=(xv.y-mean)*inv*gv.y+bv.y;
    r.z=(xv.z-mean)*inv*gv.z+bv.z;
    r.w=(xv.w-mean)*inv*gv.w+bv.w;
    op[t]=r;
  }
}

// same, but adds a per-row f32 vector (one-hot @ npd_w contribution) before LN
__device__ inline void ln_from_lds_add(const float* sC, const float* addBase, int addRow,
                                       const float* gs, const float* gb,
                                       float* outBase, int lane) {
  int row = lane & 15, half = lane >> 4;
  const float4* rp = (const float4*)(sC + row*128 + half*64);
  const float4* ap = (const float4*)(addBase + (size_t)addRow*128 + half*64);
  float s=0.f, q=0.f;
  #pragma unroll
  for (int t=0;t<16;++t){
    float4 xv = rp[t], av = ap[t];
    float x0=xv.x+av.x, x1=xv.y+av.y, x2=xv.z+av.z, x3=xv.w+av.w;
    s += x0+x1+x2+x3;
    q += x0*x0+x1*x1+x2*x2+x3*x3;
  }
  s += __shfl_xor(s,16,32); q += __shfl_xor(q,16,32);
  float mean = s*(1.0f/128.0f);
  float var  = q*(1.0f/128.0f) - mean*mean;
  float inv  = rsqrtf(var + 1e-5f);
  const float4* gp = (const float4*)(gs + half*64);
  const float4* bp = (const float4*)(gb + half*64);
  float4* op = (float4*)(outBase + (size_t)row*128 + half*64);
  #pragma unroll
  for (int t=0;t<16;++t){
    float4 xv = rp[t], av = ap[t], gv = gp[t], bv = bp[t], r;
    r.x=(xv.x+av.x-mean)*inv*gv.x+bv.x;
    r.y=(xv.y+av.y-mean)*inv*gv.y+bv.y;
    r.z=(xv.z+av.z-mean)*inv*gv.z+bv.z;
    r.w=(xv.w+av.w-mean)*inv*gv.w+bv.w;
    op[t]=r;
  }
}

// ---------------- pack f32 weights into f16 B-fragment order, (nt,ks)-major ------------
// dst[((nt*ksteps+ks)*32+lane)*16 + v] = w[K*128 + nt*16 + col], K = ks*32 + (v<8 ? (lane>>4)*8 + v : 16 + (lane>>4)*8 + v-8)
__global__ void pack_b_kernel(const float* __restrict__ w, int Kreal, int ksteps, int total,
                              _Float16* __restrict__ dst) {
  int idx = blockIdx.x*256 + threadIdx.x;
  if (idx >= total) return;
  int v    = idx & 15;
  int lane = (idx >> 4) & 31;
  int f    = idx >> 9;
  int ks   = f % ksteps;
  int nt   = f / ksteps;
  int kg = lane >> 4, col = lane & 15;
  int K = ks*32 + ((v < 8) ? (kg*8 + v) : (16 + kg*8 + (v - 8)));
  dst[idx] = (K < Kreal) ? (_Float16)w[(size_t)K*128 + nt*16 + col] : (_Float16)0.0f;
}

// ---------------- precompute LN(jn_w) for the 105 atom types ----------------
__global__ void ln_jn_kernel(const float* __restrict__ jn_w, const float* __restrict__ s,
                             const float* __restrict__ b, float* __restrict__ jnln) {
  int lane = threadIdx.x;
  const float* p = jn_w + (size_t)blockIdx.x*128;
  float* o = jnln + (size_t)blockIdx.x*128;
  float x[4]; float sum=0.f, ss=0.f;
  #pragma unroll
  for (int q=0;q<4;++q){ x[q]=p[lane+32*q]; sum+=x[q]; ss+=x[q]*x[q]; }
  #pragma unroll
  for (int mm=16;mm>0;mm>>=1){ sum += __shfl_xor(sum,mm,32); ss += __shfl_xor(ss,mm,32); }
  float mean = sum*(1.0f/128.0f);
  float var  = ss*(1.0f/128.0f) - mean*mean;
  float inv  = rsqrtf(var + 1e-5f);
  #pragma unroll
  for (int q=0;q<4;++q){ int n=lane+32*q; o[n] = (x[q]-mean)*inv*s[n] + b[n]; }
}

// ---------------- backbone: N,Ca,C,O,Cb per residue ----------------
__global__ void backbone_kernel(const float* __restrict__ X, float* __restrict__ bb) {
  int i = blockIdx.x*blockDim.x + threadIdx.x;
  if (i >= LRES) return;
  const float* x = X + (size_t)i*36*3;
  float N[3]={x[0],x[1],x[2]}, Ca[3]={x[3],x[4],x[5]}, C[3]={x[6],x[7],x[8]}, O[3]={x[9],x[10],x[11]};
  float bv[3]={Ca[0]-N[0],Ca[1]-N[1],Ca[2]-N[2]};
  float cv[3]={C[0]-Ca[0],C[1]-Ca[1],C[2]-Ca[2]};
  float av[3]={bv[1]*cv[2]-bv[2]*cv[1], bv[2]*cv[0]-bv[0]*cv[2], bv[0]*cv[1]-bv[1]*cv[0]};
  float* o = bb + (size_t)i*15;
  #pragma unroll
  for (int c3=0;c3<3;++c3){
    o[0+c3]=N[c3]; o[3+c3]=Ca[c3]; o[6+c3]=C[c3]; o[9+c3]=O[c3];
    o[12+c3] = -0.58273431f*av[c3] + 0.56802827f*bv[c3] - 0.54067466f*cv[c3] + Ca[c3];
  }
}

// ---------------- top-30 neighbors per residue ----------------
__global__ void topk30_kernel(const float* __restrict__ bb, const float* __restrict__ mask,
                              int* __restrict__ eidx, float* __restrict__ dnb,
                              float* __restrict__ eidx_out) {
  __shared__ float s_dist[LRES];
  __shared__ float s_rval[256];
  __shared__ int   s_ridx[256];
  int i = blockIdx.x, tid = threadIdx.x;
  float mi = mask[i];
  const float* cai = &bb[(size_t)i*15+3];
  float cax=cai[0], cay=cai[1], caz=cai[2];
  float lmax = -FLT_MAX;
  for (int c=tid;c<LRES;c+=256){
    const float* cc = &bb[(size_t)c*15+3];
    float dx=cax-cc[0], dy=cay-cc[1], dz=caz-cc[2];
    float d = sqrtf(dx*dx+dy*dy+dz*dz+1e-6f);
    float Dv = mi*mask[c]*d;
    s_dist[c]=Dv;
    if (Dv>lmax) lmax=Dv;
  }
  s_rval[tid]=lmax;
  __syncthreads();
  for (int off=128;off>0;off>>=1){ if (tid<off && s_rval[tid+off]>s_rval[tid]) s_rval[tid]=s_rval[tid+off]; __syncthreads(); }
  float Dmax = s_rval[0];
  __syncthreads();
  for (int c=tid;c<LRES;c+=256){
    float m2 = mi*mask[c];
    s_dist[c] += (1.0f-m2)*Dmax;
  }
  __syncthreads();
  for (int k=0;k<TOPK;++k){
    float best=FLT_MAX; int bidx=LRES;
    for (int c=tid;c<LRES;c+=256){ float v=s_dist[c]; if (v<best){best=v;bidx=c;} }
    s_rval[tid]=best; s_ridx[tid]=bidx;
    __syncthreads();
    for (int off=128;off>0;off>>=1){
      if (tid<off){
        float v2=s_rval[tid+off]; int i2=s_ridx[tid+off];
        if (v2<s_rval[tid] || (v2==s_rval[tid] && i2<s_ridx[tid])){ s_rval[tid]=v2; s_ridx[tid]=i2; }
      }
      __syncthreads();
    }
    if (tid==0){
      int sel=s_ridx[0];
      eidx[i*TOPK+k]=sel; dnb[i*TOPK+k]=s_rval[0];
      eidx_out[i*TOPK+k]=(float)sel;
      s_dist[sel]=FLT_MAX;
    }
    __syncthreads();
  }
}

// ---------------- top-25 atom context per residue ----------------
__global__ void topkj_kernel(const float* __restrict__ bb,
    const float* __restrict__ X, const float* __restrict__ X_m,
    const float* __restrict__ Y, const float* __restrict__ Y_m,
    const float* __restrict__ Z, const float* __restrict__ Z_m, const int* __restrict__ Z_t,
    const float* __restrict__ mask, const int* __restrict__ eidx,
    float* __restrict__ jp, int* __restrict__ jt,
    float* __restrict__ eidxj_out, float* __restrict__ jm_out) {
  __shared__ float s_dist[NJ];
  __shared__ float s_rval[256];
  __shared__ int   s_ridx[256];
  int i=blockIdx.x, tid=threadIdx.x;
  float mi=mask[i];
  const float* cb=&bb[(size_t)i*15+12];
  float cbx=cb[0],cby=cb[1],cbz=cb[2];
  for (int j=tid;j<NJ;j+=256){
    float p[3]; float am; int at;
    j_atom(i,j,X,X_m,Y,Y_m,Z,Z_m,Z_t,eidx,p,&am,&at);
    float dx=cbx-p[0],dy=cby-p[1],dz=cbz-p[2];
    float d=sqrtf(dx*dx+dy*dy+dz*dz+1e-6f);
    float mJ=mi*am;
    s_dist[j]=d*mJ + (1.0f-mJ)*10000.0f;
  }
  __syncthreads();
  for (int k=0;k<ACTX;++k){
    float best=FLT_MAX; int bidx=NJ;
    for (int j=tid;j<NJ;j+=256){ float v=s_dist[j]; if (v<best){best=v;bidx=j;} }
    s_rval[tid]=best; s_ridx[tid]=bidx;
    __syncthreads();
    for (int off=128;off>0;off>>=1){
      if (tid<off){
        float v2=s_rval[tid+off]; int i2=s_ridx[tid+off];
        if (v2<s_rval[tid] || (v2==s_rval[tid] && i2<s_ridx[tid])){ s_rval[tid]=v2; s_ridx[tid]=i2; }
      }
      __syncthreads();
    }
    if (tid==0){
      int sel=s_ridx[0]; float DJ=s_rval[0];
      float p[3]; float am; int at;
      j_atom(i,sel,X,X_m,Y,Y_m,Z,Z_m,Z_t,eidx,p,&am,&at);
      float mJ=mi*am;
      float far=(DJ<20.0f)?1.0f:0.0f;
      int o=i*ACTX+k;
      jp[(size_t)o*3+0]=p[0]; jp[(size_t)o*3+1]=p[1]; jp[(size_t)o*3+2]=p[2];
      jt[o]=at;
      eidxj_out[o]=(float)sel;
      jm_out[o]=mJ*far;
      s_dist[sel]=FLT_MAX;
    }
    __syncthreads();
  }
}

// ---------------- E: edge GEMM (K=416 = 13 wmma k-steps), nt-outer, fused LN ------------
__global__ void edge_gemm_kernel(const float* __restrict__ bb, const int* __restrict__ eidx,
    const float* __restrict__ dnb, const int* __restrict__ res_idx, const int* __restrict__ chain,
    const float* __restrict__ pe_w, const float* __restrict__ pe_b,
    const _Float16* __restrict__ pw, const float* __restrict__ gs, const float* __restrict__ gb,
    float* __restrict__ outE) {
  __shared__ __attribute__((aligned(32))) _Float16 sA[16][416];
  __shared__ __attribute__((aligned(16))) float sC[16*128];
  int lane = threadIdx.x;
  int tile = blockIdx.x;
  for (int idx = lane; idx < 16*26; idx += 32) {
    int e = idx/26, g = idx%26;
    int ge = tile*16 + e;
    int i = ge/TOPK;
    int nb = eidx[ge];
    _Float16* dst = &sA[e][g*16];
    if (g == 0) {
      int off = res_idx[i] - res_idx[nb];
      int dc;
      if (chain[i] == chain[nb]) { dc = off + 32; if (dc < 0) dc = 0; if (dc > 64) dc = 64; }
      else dc = 65;
      #pragma unroll
      for (int jj=0;jj<16;++jj) dst[jj] = (_Float16)(pe_w[dc*16+jj] + pe_b[jj]);
    } else if (g == 1) {
      rbf16h(dnb[ge], dst);
    } else {
      int pI = g-2;
      float d = dist3(&bb[(size_t)i*15 + c_pa[pI]*3], &bb[(size_t)nb*15 + c_pb[pI]*3]);
      rbf16h(d, dst);
    }
  }
  __syncthreads();
  int m = lane & 15, kg = lane >> 4, col = lane & 15;
  int mo = (lane < 16) ? 0 : 8;
  for (int nt=0; nt<8; ++nt) {
    v8f acc = {};
    for (int ks=0; ks<13; ++ks) {
      v16h af = load_a_frag(&sA[m][0], ks, kg);
      v16h bf = *(const v16h*)(pw + ((size_t)(nt*13+ks)*32 + lane)*16);
      acc = __builtin_amdgcn_wmma_f32_16x16x32_f16(false, af, false, bf,
                                                   (short)0, acc, false, false);
    }
    #pragma unroll
    for (int v=0; v<8; ++v) sC[(v+mo)*128 + nt*16 + col] = acc[v];
  }
  __syncthreads();
  ln_from_lds(sC, gs, gb, outE + (size_t)tile*16*128, lane);
}

// -------- V: node GEMM (K=80 pad 96), nt-outer, one-hot gather folded into fused LN -----
__global__ void v_gemm_kernel(const float* __restrict__ bb, const float* __restrict__ jp,
    const int* __restrict__ jt, const float* __restrict__ npd_w,
    const _Float16* __restrict__ pw, const float* __restrict__ gs, const float* __restrict__ gb,
    float* __restrict__ outV) {
  __shared__ __attribute__((aligned(32))) _Float16 sA[16][96];
  __shared__ __attribute__((aligned(16))) float sC[16*128];
  __shared__ int sT[16];
  int lane = threadIdx.x, tile = blockIdx.x;
  for (int idx=lane; idx<16*6; idx+=32) {
    int e = idx/6, g = idx%6;
    int r = tile*16+e; int i = r/ACTX;
    if (g < 5) {
      float d = dist3(&bb[(size_t)i*15 + g*3], &jp[(size_t)r*3]);
      rbf16h(d, &sA[e][g*16]);
    } else {
      #pragma unroll
      for (int jj=0;jj<16;++jj) sA[e][80+jj] = (_Float16)0.0f;
    }
  }
  if (lane < 16) sT[lane] = jt[tile*16+lane];
  __syncthreads();
  int m = lane&15, kg = lane>>4, col = lane&15;
  int mo = (lane < 16) ? 0 : 8;
  for (int nt=0; nt<8; ++nt) {
    v8f acc = {};
    for (int ks=0;ks<3;++ks) {
      v16h af = load_a_frag(&sA[m][0], ks, kg);
      v16h bf = *(const v16h*)(pw + ((size_t)(nt*3+ks)*32 + lane)*16);
      acc = __builtin_amdgcn_wmma_f32_16x16x32_f16(false, af, false, bf,
                                                   (short)0, acc, false, false);
    }
    #pragma unroll
    for (int v=0; v<8; ++v) sC[(v+mo)*128 + nt*16 + col] = acc[v];
  }
  __syncthreads();
  int row = lane & 15;
  ln_from_lds_add(sC, npd_w, 80 + sT[row], gs, gb, outV + (size_t)tile*16*128, lane);
}

// ---------------- J_edges: pair-RBF GEMM (K=16 pad 32), fused LN ----------------
__global__ void je_gemm_kernel(const float* __restrict__ jp, const _Float16* __restrict__ pw,
                               const float* __restrict__ gs, const float* __restrict__ gb,
                               float* __restrict__ outJE) {
  __shared__ __attribute__((aligned(32))) _Float16 sA[16][32];
  __shared__ __attribute__((aligned(16))) float sC[16*128];
  int lane=threadIdx.x, tile=blockIdx.x;
  if (lane < 16) {
    int r = tile*16+lane;
    int i = r/625, rem = r%625, a = rem/25, b = rem%25;
    float d = dist3(&jp[((size_t)i*25+a)*3], &jp[((size_t)i*25+b)*3]);
    rbf16h(d, &sA[lane][0]);
    #pragma unroll
    for (int jj=0;jj<16;++jj) sA[lane][16+jj] = (_Float16)0.0f;
  }
  __syncthreads();
  int m = lane&15, kg = lane>>4, col = lane&15;
  int mo = (lane < 16) ? 0 : 8;
  v16h af = load_a_frag(&sA[m][0], 0, kg);
  #pragma unroll
  for (int nt=0; nt<8; ++nt) {
    v8f acc = {};
    v16h bf = *(const v16h*)(pw + ((size_t)nt*32 + lane)*16);
    acc = __builtin_amdgcn_wmma_f32_16x16x32_f16(false, af, false, bf,
                                                 (short)0, acc, false, false);
    #pragma unroll
    for (int v=0; v<8; ++v) sC[(v+mo)*128 + nt*16 + col] = acc[v];
  }
  __syncthreads();
  ln_from_lds(sC, gs, gb, outJE + (size_t)tile*16*128, lane);
}

// ---------------- J_nodes: gather precomputed LN rows ----------------
__global__ void jnodes_gather_kernel(const int* __restrict__ jt, const float* __restrict__ jnln,
                                     float* __restrict__ outJN) {
  int r = blockIdx.x, lane = threadIdx.x;
  const float4* src = (const float4*)(jnln + (size_t)jt[r]*128);
  float4* dst = (float4*)(outJN + (size_t)r*128);
  dst[lane] = src[lane];
}

extern "C" void kernel_launch(void* const* d_in, const int* in_sizes, int n_in,
                              void* d_out, int out_size, void* d_ws, size_t ws_size,
                              hipStream_t stream) {
  const float* X       = (const float*)d_in[0];
  const float* X_m     = (const float*)d_in[1];
  const float* Y       = (const float*)d_in[2];
  const float* Y_m     = (const float*)d_in[3];
  const float* Z       = (const float*)d_in[4];
  const float* Z_m     = (const float*)d_in[5];
  const int*   Z_t     = (const int*)d_in[6];
  const float* mask    = (const float*)d_in[7];
  const int*   res_idx = (const int*)d_in[8];
  const int*   chain   = (const int*)d_in[9];
  const float* pe_w    = (const float*)d_in[10];
  const float* pe_b    = (const float*)d_in[11];
  const float* edge_w  = (const float*)d_in[12];
  const float* ne_s    = (const float*)d_in[13];
  const float* ne_b    = (const float*)d_in[14];
  const float* npd_w   = (const float*)d_in[15];
  const float* nn_s    = (const float*)d_in[16];
  const float* nn_b    = (const float*)d_in[17];
  const float* je_w    = (const float*)d_in[18];
  const float* nje_s   = (const float*)d_in[19];
  const float* nje_b   = (const float*)d_in[20];
  const float* jn_w    = (const float*)d_in[21];
  const float* njn_s   = (const float*)d_in[22];
  const float* njn_b   = (const float*)d_in[23];
  float* out = (float*)d_out;
  char*  ws  = (char*)d_ws;

  // workspace layout (~910 KB, all chunks 32B-aligned)
  float*     bb    = (float*)    (ws);             // 1024*15 f32
  int*       eidx  = (int*)      (ws +  61440);    // 1024*30 i32
  float*     dnb   = (float*)    (ws + 184320);    // 1024*30 f32
  float*     jp    = (float*)    (ws + 307200);    // 1024*25*3 f32
  int*       jt    = (int*)      (ws + 614400);    // 1024*25 i32
  _Float16*  pEdge = (_Float16*) (ws + 716800);    // 13*8*32*16 f16
  _Float16*  pNpd  = (_Float16*) (ws + 823296);    // 3*8*32*16 f16
  _Float16*  pJe   = (_Float16*) (ws + 847872);    // 1*8*32*16 f16
  float*     jnln  = (float*)    (ws + 856064);    // 105*128 f32

  // weight prep (independent of geometry)
  pack_b_kernel<<<208, 256, 0, stream>>>(edge_w, 416, 13, 13*8*32*16, pEdge);
  pack_b_kernel<<< 48, 256, 0, stream>>>(npd_w,   80,  3,  3*8*32*16, pNpd);
  pack_b_kernel<<< 16, 256, 0, stream>>>(je_w,    16,  1,  1*8*32*16, pJe);
  ln_jn_kernel<<<105, 32, 0, stream>>>(jn_w, njn_s, njn_b, jnln);

  backbone_kernel<<<4, 256, 0, stream>>>(X, bb);
  topk30_kernel<<<LRES, 256, 0, stream>>>(bb, mask, eidx, dnb, out + OFF_EIDX);
  topkj_kernel<<<LRES, 256, 0, stream>>>(bb, X, X_m, Y, Y_m, Z, Z_m, Z_t, mask, eidx,
                                         jp, jt, out + OFF_EIDXJ, out + OFF_JM);

  edge_gemm_kernel<<<(LRES*TOPK)/16, 32, 0, stream>>>(bb, eidx, dnb, res_idx, chain,
                                                      pe_w, pe_b, pEdge, ne_s, ne_b,
                                                      out + OFF_E);
  v_gemm_kernel<<<(LRES*ACTX)/16, 32, 0, stream>>>(bb, jp, jt, npd_w, pNpd, nn_s, nn_b,
                                                   out + OFF_V);
  je_gemm_kernel<<<(LRES*ACTX*ACTX)/16, 32, 0, stream>>>(jp, pJe, nje_s, nje_b,
                                                         out + OFF_JE);
  jnodes_gather_kernel<<<LRES*ACTX, 32, 0, stream>>>(jt, jnln, out + OFF_JN);
}
